// EvLowLightNet_34574486733125
// MI455X (gfx1250) — matrix operand order repaired and checked by reading, not compiled, not run on hardware
//
#include <hip/hip_runtime.h>

typedef __attribute__((ext_vector_type(16))) _Float16 v16h;
typedef __attribute__((ext_vector_type(8)))  float    v8f;

// ---------------------------------------------------------------------------
// Implicit-GEMM 3x3 conv (pad=1, stride 1 or 2) via V_WMMA_F32_16X16X32_F16.
// One wave per block. N-tile = 16 consecutive output pixels in a row,
// M = output channels (tiled by 16), K = CIN*9 padded to KPAD = KC*32.
// Stages: (1) f32 patch -> LDS, (2) f16 [m][k] weights + f16 im2col B
// -> LDS (zero padded), (3) branch-free WMMA loop with v16h LDS loads.
// FIRST layer gathers the virtual concat( nearest_resize(lqs,256), evs ).
// ---------------------------------------------------------------------------
template<int CIN, int COUT, int OUT, int STRIDE, bool FIRST, bool RELU, bool BIAS>
__global__ __launch_bounds__(32) void conv3x3_wmma(
    const float* __restrict__ in, const float* __restrict__ lqs,
    const float* __restrict__ evs, const float* __restrict__ w,
    const float* __restrict__ bias, float* __restrict__ out)
{
  const int MT    = (COUT + 15) / 16;
  const int NXT   = OUT / 16;
  const int IN_HW = OUT * STRIDE;
  const int PW    = 16 * STRIDE + 2;   // staged patch width
  const int K     = CIN * 9;
  const int KC    = (K + 31) / 32;
  const int KPAD  = KC * 32;

  __shared__ float patch[CIN * 3 * PW];
  __shared__ __align__(32) _Float16 wA[16 * KPAD];    // [m][k] natural order
  __shared__ __align__(32) _Float16 bcol[16 * KPAD];  // [n][k] natural order

  int task  = blockIdx.x;
  int mtile = task % MT;  task /= MT;
  int xt    = task % NXT; task /= NXT;
  int y     = task % OUT; task /= OUT;
  int b     = task;
  int x0    = xt * 16;
  int lane  = threadIdx.x;
  int mrowbase = mtile * 16;

  // ---- stage 1: input patch (CIN x 3 x PW) f32, zero padding at borders ----
  for (int i = lane; i < CIN * 3 * PW; i += 32) {
    int c = i / (3 * PW); int rem = i % (3 * PW);
    int ky = rem / PW;    int lx = rem % PW;
    int iy = STRIDE * y  - 1 + ky;
    int ix = STRIDE * x0 - 1 + lx;
    float v = 0.f;
    if (iy >= 0 && iy < IN_HW && ix >= 0 && ix < IN_HW) {
      if (FIRST) {
        v = (c < 3) ? lqs[((b * 3 + c) * 1024 + iy * 4) * 1024 + ix * 4]
                    : evs[((b * 15 + (c - 3)) * 256 + iy) * 256 + ix];
      } else {
        v = in[((b * CIN + c) * IN_HW + iy) * IN_HW + ix];
      }
    }
    patch[i] = v;
  }
  // ---- stage 1b: weights -> f16 LDS in natural [m][k] order ---------------
  for (int i = lane; i < 16 * KPAD; i += 32) {
    int m = i / KPAD, k = i % KPAD;
    float v = 0.f;
    if (k < K && (mrowbase + m) < COUT) v = w[(mrowbase + m) * K + k];
    int sw = k;
    wA[m * KPAD + sw] = (_Float16)v;
  }
  __syncthreads();

  // ---- stage 2: im2col expansion patch -> bcol f16 (zero padded) ----------
  for (int i = lane; i < 16 * KPAD; i += 32) {
    int n = i / KPAD, k = i % KPAD;
    float v = 0.f;
    if (k < K) {
      int c = k / 9, r2 = k % 9, ky = r2 / 3, kx = r2 % 3;
      v = patch[(c * 3 + ky) * PW + STRIDE * n + kx];
    }
    bcol[i] = (_Float16)v;   // i == n*KPAD + k
  }
  __syncthreads();

  // ---- stage 3: branch-free WMMA loop -------------------------------------
  int ln = lane & 15;   // A: row M; B: col N (per ISA wave32 layouts)
  int hi = lane >> 4;
  v8f acc = {};
#pragma unroll
  for (int kc = 0; kc < KC; ++kc) {
    v16h afrag = *reinterpret_cast<const v16h*>(wA   + ln * KPAD + kc * 32 + 16 * hi);
    v16h bfrag = *reinterpret_cast<const v16h*>(bcol + ln * KPAD + kc * 32 + 16 * hi);
    acc = __builtin_amdgcn_wmma_f32_16x16x32_f16(
        false, afrag, false, bfrag, (short)0, acc, false, false);
  }

  // D layout: lane holds column N = ln; VGPR v holds row M = v + 8*hi
#pragma unroll
  for (int v = 0; v < 8; ++v) {
    int mrow = mrowbase + v + 8 * hi;
    if (mrow < COUT) {
      float val = acc[v];
      if (BIAS) val += bias[mrow];
      if (RELU) val = fmaxf(val, 0.f);
      out[((b * COUT + mrow) * OUT + y) * OUT + x0 + ln] = val;
    }
  }
}

// ---------------------------------------------------------------------------
// Small scalar conv (3x3, pad 1, stride 2, relu) for the global head (8x8/4x4)
// ---------------------------------------------------------------------------
template<int CIN, int COUT, int OUT, int STRIDE>
__global__ void conv_small(const float* __restrict__ in, const float* __restrict__ w,
                           const float* __restrict__ bias, float* __restrict__ out, int NB)
{
  const int IN_HW = OUT * STRIDE;
  int idx = blockIdx.x * blockDim.x + threadIdx.x;
  int total = NB * COUT * OUT * OUT;
  if (idx >= total) return;
  int x = idx % OUT; int t = idx / OUT;
  int y = t % OUT;   t /= OUT;
  int o = t % COUT;  int b = t / COUT;
  float acc = bias[o];
  for (int c = 0; c < CIN; ++c)
    for (int ky = 0; ky < 3; ++ky) {
      int iy = STRIDE * y - 1 + ky; if (iy < 0 || iy >= IN_HW) continue;
      for (int kx = 0; kx < 3; ++kx) {
        int ix = STRIDE * x - 1 + kx; if (ix < 0 || ix >= IN_HW) continue;
        acc += w[((o * CIN + c) * 3 + ky) * 3 + kx]
             * in[((b * CIN + c) * IN_HW + iy) * IN_HW + ix];
      }
    }
  out[idx] = fmaxf(acc, 0.f);
}

// ---------------------------------------------------------------------------
__global__ void fc_kernel(const float* __restrict__ x, const float* __restrict__ w,
                          const float* __restrict__ bias, float* __restrict__ y,
                          int NB, int IN, int OUT, int relu)
{
  int idx = blockIdx.x * blockDim.x + threadIdx.x;
  if (idx >= NB * OUT) return;
  int o = idx % OUT, b = idx / OUT;
  float acc = bias[o];
  const float* xr = x + b * IN;
  const float* wr = w + o * IN;
  for (int k = 0; k < IN; ++k) acc += xr[k] * wr[k];
  y[idx] = relu ? fmaxf(acc, 0.f) : acc;
}

// fused = relu(xg + xl); bg = fuw(1x1) * fused + fub  -> [B,8,16,16]
__global__ void fuse_grid(const float* __restrict__ xg, const float* __restrict__ xl,
                          const float* __restrict__ fuw, const float* __restrict__ fub,
                          float* __restrict__ bg, int NB)
{
  int idx = blockIdx.x * blockDim.x + threadIdx.x;
  if (idx >= NB * 8 * 256) return;
  int px = idx % 256; int t = idx / 256;
  int d = t % 8; int b = t / 8;
  float acc = fub[d];
  for (int ch = 0; ch < 64; ++ch) {
    float f = xg[b * 64 + ch] + xl[(b * 64 + ch) * 256 + px];
    acc += fuw[d * 64 + ch] * fmaxf(f, 0.f);
  }
  bg[(b * 8 + d) * 256 + px] = acc;
}

// ---------------------------------------------------------------------------
// Fused high-res pass: 17x17 separable Gaussian (reflect pad) -> per-pixel
// guide (1x1 convs 3->16->1, sigmoid*2-0.5) -> trilinear slice of the
// bilateral grid (LDS resident). One read of lqs, one write of out.
// Tile: 32x32 pixels per 256-thread block (4 px/thread). LDS ~54 KB.
// ---------------------------------------------------------------------------
__global__ __launch_bounds__(256) void highres_slice(
    const float* __restrict__ lqs, const float* __restrict__ gw1,
    const float* __restrict__ gb1, const float* __restrict__ gw2,
    const float* __restrict__ gb2, const float* __restrict__ bg,
    float* __restrict__ out)
{
  const int H = 1024, W = 1024;
  __shared__ float s_in[3 * 48 * 48];
  __shared__ float s_hb[3 * 48 * 32];
  __shared__ float s_bg[8 * 16 * 16];
  __shared__ float s_w[81];

  int tid = threadIdx.x;
  int x0 = blockIdx.x * 32, y0 = blockIdx.y * 32, b = blockIdx.z;

  if (tid < 48)       s_w[tid] = gw1[tid];
  else if (tid < 64)  s_w[tid] = gb1[tid - 48];
  else if (tid < 80)  s_w[tid] = gw2[tid - 64];
  else if (tid == 80) s_w[80]  = gb2[0];

  for (int i = tid; i < 2048; i += 256) s_bg[i] = bg[b * 2048 + i];

  for (int i = tid; i < 3 * 48 * 48; i += 256) {
    int c = i / (48 * 48); int r = (i / 48) % 48; int cc = i % 48;
    int iy = y0 - 8 + r; int ix = x0 - 8 + cc;
    iy = iy < 0 ? -iy : (iy >= H ? 2 * H - 2 - iy : iy);   // reflect
    ix = ix < 0 ? -ix : (ix >= W ? 2 * W - 2 - ix : ix);
    s_in[i] = lqs[((b * 3 + c) * H + iy) * W + ix];
  }
  __syncthreads();

  float g[17];
  {
    float s = 0.f;
    for (int i = 0; i < 17; ++i) { float t = (float)i - 8.f; g[i] = __expf(-t * t * 0.125f); s += g[i]; }
    float inv = 1.f / s;
    for (int i = 0; i < 17; ++i) g[i] *= inv;
  }

  // horizontal blur: 48 rows x 32 cols x 3 ch
  for (int i = tid; i < 3 * 48 * 32; i += 256) {
    int c = i / (48 * 32); int r = (i / 32) % 48; int cx = i % 32;
    const float* row = &s_in[(c * 48 + r) * 48 + cx];
    float acc = 0.f;
#pragma unroll
    for (int k = 0; k < 17; ++k) acc += g[k] * row[k];
    s_hb[(c * 48 + r) * 32 + cx] = acc;
  }
  __syncthreads();

  int lx = tid & 31, ly0 = tid >> 5;
#pragma unroll
  for (int p = 0; p < 4; ++p) {
    int ly = ly0 + p * 8;
    float rgb[3];
#pragma unroll
    for (int c = 0; c < 3; ++c) {
      float acc = 0.f;
#pragma unroll
      for (int k = 0; k < 17; ++k) acc += g[k] * s_hb[(c * 48 + ly + k) * 32 + lx];
      rgb[c] = acc;
    }
    float a2 = s_w[80];
#pragma unroll
    for (int o = 0; o < 16; ++o) {
      float h = s_w[48 + o] + s_w[o * 3] * rgb[0] + s_w[o * 3 + 1] * rgb[1] + s_w[o * 3 + 2] * rgb[2];
      a2 += s_w[64 + o] * fmaxf(h, 0.f);
    }
    float guide = 2.f / (1.f + __expf(-a2)) - 0.5f;

    int X = x0 + lx, Y = y0 + ly;
    float gxn = (X + 0.5f) * (2.f / 1024.f) - 0.5f;
    float gyn = (Y + 0.5f) * (2.f / 1024.f) - 0.5f;
    float xs = fminf(fmaxf(((gxn + 1.f) * 16.f - 1.f) * 0.5f, 0.f), 15.f);
    float ys = fminf(fmaxf(((gyn + 1.f) * 16.f - 1.f) * 0.5f, 0.f), 15.f);
    float zs = fminf(fmaxf(((guide + 1.f) * 8.f - 1.f) * 0.5f, 0.f), 7.f);
    float xf = floorf(xs), yf = floorf(ys), zf = floorf(zs);
    int xi = (int)xf, yi = (int)yf, zi = (int)zf;
    float fx = xs - xf, fy = ys - yf, fz = zs - zf;
    int xj = xi + 1 < 15 ? xi + 1 : 15;
    int yj = yi + 1 < 15 ? yi + 1 : 15;
    int zj = zi + 1 < 7  ? zi + 1 : 7;
    auto G = [&](int z, int y, int x) { return s_bg[(z * 16 + y) * 16 + x]; };
    float v =
        G(zi, yi, xi) * (1 - fz) * (1 - fy) * (1 - fx) + G(zi, yi, xj) * (1 - fz) * (1 - fy) * fx +
        G(zi, yj, xi) * (1 - fz) * fy * (1 - fx)       + G(zi, yj, xj) * (1 - fz) * fy * fx +
        G(zj, yi, xi) * fz * (1 - fy) * (1 - fx)       + G(zj, yi, xj) * fz * (1 - fy) * fx +
        G(zj, yj, xi) * fz * fy * (1 - fx)             + G(zj, yj, xj) * fz * fy * fx;
    out[(b * H + Y) * W + X] = v;
  }
}

// ---------------------------------------------------------------------------
extern "C" void kernel_launch(void* const* d_in, const int* in_sizes, int n_in,
                              void* d_out, int out_size, void* d_ws, size_t ws_size,
                              hipStream_t stream)
{
  (void)n_in; (void)out_size; (void)ws_size;
  const float* lqs = (const float*)d_in[0];
  const float* evs = (const float*)d_in[1];
  const float* gw1 = (const float*)d_in[2];
  const float* gb1 = (const float*)d_in[3];
  const float* gw2 = (const float*)d_in[4];
  const float* gb2 = (const float*)d_in[5];
  const float* sw0 = (const float*)d_in[6];
  const float* sb0 = (const float*)d_in[7];
  const float* sw1 = (const float*)d_in[8];
  const float* sb1 = (const float*)d_in[9];
  const float* sw2 = (const float*)d_in[10];
  const float* sb2 = (const float*)d_in[11];
  const float* sw3 = (const float*)d_in[12];
  const float* sb3 = (const float*)d_in[13];
  const float* cw0 = (const float*)d_in[14];
  const float* cb0 = (const float*)d_in[15];
  const float* cw1 = (const float*)d_in[16];
  const float* cb1 = (const float*)d_in[17];
  const float* fw1 = (const float*)d_in[18];
  const float* fb1 = (const float*)d_in[19];
  const float* fw2 = (const float*)d_in[20];
  const float* fb2 = (const float*)d_in[21];
  const float* fw3 = (const float*)d_in[22];
  const float* fb3 = (const float*)d_in[23];
  const float* lw1 = (const float*)d_in[24];
  const float* lb1 = (const float*)d_in[25];
  const float* lw2 = (const float*)d_in[26];
  const float* fuw = (const float*)d_in[27];
  const float* fub = (const float*)d_in[28];
  float* out = (float*)d_out;
  float* ws  = (float*)d_ws;

  int B = in_sizes[0] / (3 * 1024 * 1024);

  float* act1  = ws;                           // [B,8,128,128]
  float* act2  = act1  + (size_t)B * 8  * 128 * 128;
  float* act3  = act2  + (size_t)B * 16 * 64 * 64;
  float* act4  = act3  + (size_t)B * 32 * 32 * 32;
  float* loc1  = act4  + (size_t)B * 64 * 16 * 16;
  float* xl    = loc1  + (size_t)B * 64 * 16 * 16;
  float* g1    = xl    + (size_t)B * 64 * 16 * 16;
  float* g2    = g1    + (size_t)B * 64 * 8 * 8;
  float* f1    = g2    + (size_t)B * 64 * 4 * 4;
  float* f2    = f1    + (size_t)B * 256;
  float* xg    = f2    + (size_t)B * 128;
  float* bgrid = xg    + (size_t)B * 64;

  dim3 w32(32);
  // low-res stream (WMMA implicit GEMM convs)
  conv3x3_wmma<18, 8, 128, 2, true,  true,  true ><<<B * 128 * 8 * 1, w32, 0, stream>>>(nullptr, lqs, evs, sw0, sb0, act1);
  conv3x3_wmma< 8, 16,  64, 2, false, true,  true ><<<B *  64 * 4 * 1, w32, 0, stream>>>(act1, nullptr, nullptr, sw1, sb1, act2);
  conv3x3_wmma<16, 32,  32, 2, false, true,  true ><<<B *  32 * 2 * 2, w32, 0, stream>>>(act2, nullptr, nullptr, sw2, sb2, act3);
  conv3x3_wmma<32, 64,  16, 2, false, true,  true ><<<B *  16 * 1 * 4, w32, 0, stream>>>(act3, nullptr, nullptr, sw3, sb3, act4);
  // local head
  conv3x3_wmma<64, 64,  16, 1, false, true,  true ><<<B *  16 * 1 * 4, w32, 0, stream>>>(act4, nullptr, nullptr, lw1, lb1, loc1);
  conv3x3_wmma<64, 64,  16, 1, false, false, false><<<B *  16 * 1 * 4, w32, 0, stream>>>(loc1, nullptr, nullptr, lw2, nullptr, xl);
  // global head
  { int tot = B * 64 * 8 * 8; conv_small<64, 64, 8, 2><<<(tot + 255) / 256, 256, 0, stream>>>(act4, cw0, cb0, g1, B); }
  { int tot = B * 64 * 4 * 4; conv_small<64, 64, 4, 2><<<(tot + 255) / 256, 256, 0, stream>>>(g1, cw1, cb1, g2, B); }
  fc_kernel<<<(B * 256 + 255) / 256, 256, 0, stream>>>(g2, fw1, fb1, f1, B, 1024, 256, 1);
  fc_kernel<<<(B * 128 + 255) / 256, 256, 0, stream>>>(f1, fw2, fb2, f2, B, 256, 128, 1);
  fc_kernel<<<(B * 64  + 255) / 256, 256, 0, stream>>>(f2, fw3, fb3, xg, B, 128, 64, 0);
  // bilateral grid
  fuse_grid<<<(B * 8 * 256 + 255) / 256, 256, 0, stream>>>(xg, xl, fuw, fub, bgrid, B);
  // fused high-res: blur -> guide -> slice (single pass over lqs)
  highres_slice<<<dim3(32, 32, B), dim3(256), 0, stream>>>(lqs, gw1, gb1, gw2, gb2, bgrid, out);
}
